// GeneratorDecoderBlock_18064632447047
// MI455X (gfx1250) — matrix operand order repaired and checked by reading, not compiled, not run on hardware
//
#include <hip/hip_runtime.h>
#include <hip/hip_bf16.h>

// ---------------------------------------------------------------------------
// StyleGAN2 generator block on MI455X (gfx1250).
// Strategy: fold modulation into the input (per-channel scale -> bf16),
// fold demodulation into the epilogue (per-(b,oc) rsqrt scale), so both 3x3
// convs become plain implicit GEMMs with BATCH-INVARIANT bf16 weights that we
// pre-pack into a v_wmma_f32_16x16x32_bf16 A-fragment lane layout (contiguous k).
// Compute-bound: ~620 GFLOP vs ~0.5 GB of traffic (all L2-resident).
// ---------------------------------------------------------------------------

typedef __attribute__((ext_vector_type(16))) __bf16 v16bf;
typedef __attribute__((ext_vector_type(8)))  __bf16 v8bf;
typedef __attribute__((ext_vector_type(8)))  float  v8f;

#define B_   16
#define C_   512
#define D_   512
#define HW_  4096

// 1/sqrt(512), 1/sqrt(512*9)
#define C_STYLE 0.044194173824159216f
#define C_CONV  0.014731391274719738f

__device__ __forceinline__ float wave_reduce(float v) {
  v += __shfl_xor(v, 16, 32);
  v += __shfl_xor(v, 8, 32);
  v += __shfl_xor(v, 4, 32);
  v += __shfl_xor(v, 2, 32);
  v += __shfl_xor(v, 1, 32);
  return v;
}

// --------------------------- style projection -----------------------------
// s[layer][b][oc] = dot(w[b,:], style_w[oc,:]) * c + style_b[oc]
// one wave per output; 3 layers * 16 * 512 = 24576 dots, 8 waves/block.
__global__ void style_kernel(const float* __restrict__ w,
                             const float* __restrict__ sw1, const float* __restrict__ sb1,
                             const float* __restrict__ sw2, const float* __restrict__ sb2,
                             const float* __restrict__ swr, const float* __restrict__ sbr,
                             float* __restrict__ s1, float* __restrict__ s2,
                             float* __restrict__ sr) {
  const int wv = threadIdx.x >> 5, lane = threadIdx.x & 31;
  int id = blockIdx.x * 8 + wv;            // 0..24575
  int layer = id >> 13;
  int rem = id & 8191;
  int b = rem >> 9, oc = rem & 511;
  const float* sw = (layer == 0) ? sw1 : (layer == 1 ? sw2 : swr);
  const float* sb = (layer == 0) ? sb1 : (layer == 1 ? sb2 : sbr);
  float*       so = (layer == 0) ? s1  : (layer == 1 ? s2  : sr);
  const float* wrow = w + (size_t)b * D_;
  const float* srow = sw + (size_t)oc * D_;
  float acc = 0.f;
#pragma unroll
  for (int j = 0; j < 16; ++j) {
    int k = lane + j * 32;
    acc += wrow[k] * srow[k];
  }
  acc = wave_reduce(acc);
  if (lane == 0) so[b * C_ + oc] = acc * C_STYLE + sb[oc];
}

// ------------------------------ demod scale --------------------------------
// sig[b][oc] = rsqrt( c^2 * sum_ic s[b,ic]^2 * sum_kk w[oc,ic,kk]^2 + eps )
__global__ void sigma_kernel(const float* __restrict__ s1, const float* __restrict__ s2,
                             const float* __restrict__ cw1, const float* __restrict__ cw2,
                             float* __restrict__ sig1, float* __restrict__ sig2) {
  const int wv = threadIdx.x >> 5, lane = threadIdx.x & 31;
  int id = blockIdx.x * 8 + wv;            // 0..16383
  int conv = id >> 13;
  int rem = id & 8191;
  int b = rem >> 9, oc = rem & 511;
  const float* s  = conv ? s2  : s1;
  const float* cw = conv ? cw2 : cw1;
  float acc = 0.f;
  for (int j = 0; j < 16; ++j) {
    int ic = lane + j * 32;
    float sv = s[b * C_ + ic];
    const float* wp = cw + (size_t)(oc * C_ + ic) * 9;
    float w2 = 0.f;
#pragma unroll
    for (int kk = 0; kk < 9; ++kk) { float xv = wp[kk]; w2 += xv * xv; }
    acc += sv * sv * w2;
  }
  acc = wave_reduce(acc);
  if (lane == 0) {
    const float c2 = C_CONV * C_CONV;
    (conv ? sig2 : sig1)[b * C_ + oc] = rsqrtf(acc * c2 + 1e-8f);
  }
}

// --------------------------- weight pre-pack -------------------------------
// Packs conv weights * c into bf16 in this code's 16x16x32 bf16 A-fragment
// order: [kidx(9)][icc(16)][oc_tile(32)][lane(32)][16 halfs].
// lane<16: K = {0..15}; lane>=16: K = {16..31}; M = lane&15.
__global__ void pack_kernel(const float* __restrict__ cw, __bf16* __restrict__ apk) {
  int P = blockIdx.x * 256 + threadIdx.x;      // < 2359296
  int e    =  P        & 15;
  int lane = (P >> 4)  & 31;
  int octg = (P >> 9)  & 31;
  int icc  = (P >> 14) & 15;
  int kidx =  P >> 18;
  int m = lane & 15, hi = lane >> 4;
  int oc = octg * 16 + m;
  int kl = e + hi * 16;
  int ic = icc * 32 + kl;
  int ky = kidx / 3, kx = kidx % 3;
  float v = cw[((size_t)(oc * C_ + ic) * 3 + ky) * 3 + kx] * C_CONV;
  apk[P] = (__bf16)v;
}

// --------------------------- implicit-GEMM conv ----------------------------
// Block: 128 oc x 128 px (2 image rows) per sample. 8 waves = 4(m) x 2(n);
// wave: 32 oc x 64 px = 2x4 WMMA tiles. K loop: 16 ic-chunks x 9 taps.
// LDS tile: 4 halo rows x 66 cols x 32 ch, stored [row][col][ic] (ic contig,
// stride 40 halfs = conflict-free, 16B-aligned B-fragment ds_load_b128).
__global__ void __launch_bounds__(256) conv_kernel(
    const float* __restrict__ xin, const float* __restrict__ s,
    const float* __restrict__ sig, const v16bf* __restrict__ apk,
    const float* __restrict__ noise, const float* __restrict__ nscale,
    const float* __restrict__ bias, float* __restrict__ out) {
  __shared__ __bf16 lds[4 * 66 * 40];        // 21120 B

  const int y0  = blockIdx.x * 2;            // first of 2 output rows
  const int ocb = blockIdx.y;                // 0..3 (128-oc block)
  const int b   = blockIdx.z;                // 0..15
  const int t = threadIdx.x;
  const int lane = t & 31, wv = t >> 5;
  const int wm = wv >> 1, wn = wv & 1;       // wn picks output row
  const int hl = lane >> 4, ll = lane & 15;

  v8f acc[2][4];
#pragma unroll
  for (int i = 0; i < 2; ++i)
#pragma unroll
    for (int j = 0; j < 4; ++j) acc[i][j] = (v8f){0.f,0.f,0.f,0.f,0.f,0.f,0.f,0.f};

  for (int icc = 0; icc < 16; ++icc) {
    __syncthreads();
    // ---- stage modulated bf16 input halo tile (4 x 66 x 32) ----
    for (int idx = t; idx < 4 * 32 * 66; idx += 256) {
      int col = idx % 66;
      int ric = idx / 66;
      int ic  = ric & 31;
      int r   = ric >> 5;                    // halo row 0..3
      int gy = y0 - 1 + r;
      int gx = col - 1;
      int icg = icc * 32 + ic;
      float v = 0.f;
      if ((unsigned)gy < 64u && (unsigned)gx < 64u)
        v = xin[(((size_t)b * C_ + icg) * 64 + gy) * 64 + gx] * s[b * C_ + icg];
      lds[(r * 66 + col) * 40 + ic] = (__bf16)v;
    }
    __syncthreads();

    const int octg0 = ocb * 8 + wm * 2;
#pragma unroll
    for (int kidx = 0; kidx < 9; ++kidx) {
      const int ky = kidx / 3, kx = kidx % 3;
      // A fragments: pre-packed, L2 resident, one 32B load per lane each
      v16bf A0 = apk[(size_t)(((kidx * 16 + icc) * 32 + octg0) * 32) + lane];
      v16bf A1 = apk[(size_t)(((kidx * 16 + icc) * 32 + octg0 + 1) * 32) + lane];
      // B fragments from LDS: lane = pixel, 16 contiguous channels per half
      v16bf Bf[4];
#pragma unroll
      for (int j = 0; j < 4; ++j) {
        int base = ((wn + ky) * 66 + j * 16 + ll + kx) * 40 + hl * 16;
        union { v16bf v; v8bf h[2]; } u;
        u.h[0] = *(const v8bf*)&lds[base];
        u.h[1] = *(const v8bf*)&lds[base + 8];
        Bf[j] = u.v;
      }
#pragma unroll
      for (int j = 0; j < 4; ++j) {
        acc[0][j] = __builtin_amdgcn_wmma_f32_16x16x32_bf16(
            false, A0, false, Bf[j], (short)0, acc[0][j], false, false);
        acc[1][j] = __builtin_amdgcn_wmma_f32_16x16x32_bf16(
            false, A1, false, Bf[j], (short)0, acc[1][j], false, false);
      }
    }
  }

  // ---- epilogue: demod scale + noise + bias + leaky relu ----
  const float ns = nscale[0];
  const int y = y0 + wn;
#pragma unroll
  for (int i = 0; i < 2; ++i) {
#pragma unroll
    for (int j = 0; j < 4; ++j) {
      int x = j * 16 + ll;
      float nz = ns * noise[b * HW_ + y * 64 + x];
      int ocbase = ocb * 128 + (wm * 2 + i) * 16 + hl * 8;
#pragma unroll
      for (int r = 0; r < 8; ++r) {
        int oc = ocbase + r;
        float v = acc[i][j][r] * sig[b * C_ + oc] + nz + bias[oc];
        v = (v >= 0.f) ? v : 0.2f * v;
        out[(((size_t)b * C_ + oc) * 64 + y) * 64 + x] = v;
      }
    }
  }
}

// ------------------------------ 1x1 to-RGB ---------------------------------
// rgb[b,o,y,x] = lrelu( c * sum_ic rw[o,ic]*s[b,ic]*x2[b,ic,y,x] + rb[o] )
// memory-bound (one pass over 134MB); block = (y,b), 4 threads share a pixel.
__global__ void rgb_kernel(const float* __restrict__ x2, const float* __restrict__ srgb,
                           const float* __restrict__ rw, const float* __restrict__ rb,
                           float* __restrict__ rgb) {
  __shared__ float red[4][64][3];
  const int y = blockIdx.x, b = blockIdx.y;
  const int t = threadIdx.x;
  const int x = t & 63, q = t >> 6;
  float a0 = 0.f, a1 = 0.f, a2 = 0.f;
  for (int ic = q * 128; ic < q * 128 + 128; ++ic) {
    float m = x2[(((size_t)b * C_ + ic) * 64 + y) * 64 + x] * srgb[b * C_ + ic];
    a0 += m * rw[ic];
    a1 += m * rw[C_ + ic];
    a2 += m * rw[2 * C_ + ic];
  }
  red[q][x][0] = a0; red[q][x][1] = a1; red[q][x][2] = a2;
  __syncthreads();
  if (q == 0) {
#pragma unroll
    for (int o = 0; o < 3; ++o) {
      float v = (red[0][x][o] + red[1][x][o] + red[2][x][o] + red[3][x][o])
                    * C_STYLE + rb[o];
      v = (v >= 0.f) ? v : 0.2f * v;
      rgb[(((size_t)b * 3 + o) * 64 + y) * 64 + x] = v;
    }
  }
}

// ------------------------------- launcher ----------------------------------
extern "C" void kernel_launch(void* const* d_in, const int* in_sizes, int n_in,
                              void* d_out, int out_size, void* d_ws, size_t ws_size,
                              hipStream_t stream) {
  const float* x   = (const float*)d_in[0];
  const float* w   = (const float*)d_in[1];
  const float* n1  = (const float*)d_in[2];
  const float* n2  = (const float*)d_in[3];
  const float* sw1 = (const float*)d_in[4];
  const float* sb1 = (const float*)d_in[5];
  const float* cw1 = (const float*)d_in[6];
  const float* ns1 = (const float*)d_in[7];
  const float* bb1 = (const float*)d_in[8];
  const float* sw2 = (const float*)d_in[9];
  const float* sb2 = (const float*)d_in[10];
  const float* cw2 = (const float*)d_in[11];
  const float* ns2 = (const float*)d_in[12];
  const float* bb2 = (const float*)d_in[13];
  const float* swr = (const float*)d_in[14];
  const float* sbr = (const float*)d_in[15];
  const float* rw  = (const float*)d_in[16];
  const float* rb  = (const float*)d_in[17];

  // workspace layout (bytes): s1,s2,srgb,sig1,sig2, Apk1, Apk2, x1
  char* ws = (char*)d_ws;
  float*  s1   = (float*)(ws + 0);
  float*  s2   = (float*)(ws + 32768);
  float*  sr   = (float*)(ws + 65536);
  float*  sg1  = (float*)(ws + 98304);
  float*  sg2  = (float*)(ws + 131072);
  __bf16* apk1 = (__bf16*)(ws + 163840);
  __bf16* apk2 = (__bf16*)(ws + 163840 + 4718592);     // 4,882,432
  float*  x1   = (float*)(ws + 9601024);               // 134,217,728 B

  float* xout = (float*)d_out;                 // 16*512*64*64 floats
  float* rgb  = (float*)d_out + 33554432;      // then 16*3*64*64 floats

  style_kernel<<<3072, 256, 0, stream>>>(w, sw1, sb1, sw2, sb2, swr, sbr, s1, s2, sr);
  sigma_kernel<<<2048, 256, 0, stream>>>(s1, s2, cw1, cw2, sg1, sg2);
  pack_kernel<<<9216, 256, 0, stream>>>(cw1, apk1);
  pack_kernel<<<9216, 256, 0, stream>>>(cw2, apk2);

  conv_kernel<<<dim3(32, 4, 16), 256, 0, stream>>>(
      x,  s1, sg1, (const v16bf*)apk1, n1, ns1, bb1, x1);
  conv_kernel<<<dim3(32, 4, 16), 256, 0, stream>>>(
      x1, s2, sg2, (const v16bf*)apk2, n2, ns2, bb2, xout);

  rgb_kernel<<<dim3(64, 16), 256, 0, stream>>>(xout, sr, rw, rb, rgb);
}
